// HeteroGCNEncoder_68118181315022
// MI455X (gfx1250) — compile-verified, matrix operand order checked by reading, not run on hardware
//
#include <hip/hip_runtime.h>
#include <hip/hip_bf16.h>
#include <stdint.h>

// ---------------- problem constants (match reference) ----------------
constexpr int NA  = 50000;
constexpr int NB  = 50000;
constexpr int E   = 800000;
constexpr int H   = 256;    // D == H; also the GEMM K dimension (fixed)
constexpr int EMB = 128;
constexpr int L   = 2;
constexpr int KK  = 256;    // GEMM reduction dim (== H) baked into kernels

// ---------------- vector types ----------------
typedef __attribute__((ext_vector_type(8)))  float          v8f;
typedef __attribute__((ext_vector_type(16))) __bf16         v16bf;
typedef __attribute__((ext_vector_type(16))) unsigned short v16u;
typedef __attribute__((ext_vector_type(8)))  unsigned short u16x8;

// f32 -> bf16 (round to nearest even)
__device__ __forceinline__ unsigned short f2bf(float f) {
    unsigned u = __float_as_uint(f);
    unsigned r = u + 0x7FFFu + ((u >> 16) & 1u);
    return (unsigned short)(r >> 16);
}
__device__ __forceinline__ float bf2f(unsigned short b) {
    return __uint_as_float(((unsigned)b) << 16);
}

// Load one 16-element bf16 fragment row-chunk: two aligned 16B loads.
// Per ISA layout (16-bit A 16x32): lanes 0-15 hold K=[k..k+8) and [k+16..k+24),
// lanes 16-31 hold K=[k+8..k+16) and [k+24..k+32); caller bakes the +0/+8
// lane offset into `p`. Works for both global and LDS pointers (the compiler
// selects global_load_b128 or ds_load_b128 after address-space inference).
template <typename P>
__device__ __forceinline__ v16bf frag16(const P* p) {
    u16x8 lo = *(const u16x8*)(p);
    u16x8 hi = *(const u16x8*)(p + 16);
    v16u v = __builtin_shufflevector(lo, hi, 0,1,2,3,4,5,6,7,8,9,10,11,12,13,14,15);
    union { v16u u; v16bf b; } cvt;
    cvt.u = v;
    return cvt.b;
}

// ---------------- kernels ----------------

// Convert f32 activations to bf16
__global__ void cvt_f32_bf16(const float* __restrict__ in,
                             unsigned short* __restrict__ out, int n) {
    int i = blockIdx.x * blockDim.x + threadIdx.x;
    if (i < n) out[i] = f2bf(in[i]);
}

// Prepare per-(layer,type) transposed bf16 weights:
//   WlT[l][t][n*H+k]  = Wl[l, rn, k, n]
//   WcT[l][t][n*H+k]  = Wr[l, rn, k, n] + Wl[l, rs, k, n] + Wr[l, rs, k, n]
//   biasc[l][t][n]    = bl[l, rn, n] + bl[l, rs, n]
// type t=0 (a): rn=1 (b,to,a), rs=2 ; type t=1 (b): rn=0 (a,to,b), rs=3
__global__ void prep_weights(const float* __restrict__ Wl,
                             const float* __restrict__ bl,
                             const float* __restrict__ Wr,
                             unsigned short* __restrict__ WlT,
                             unsigned short* __restrict__ WcT,
                             float* __restrict__ biasc) {
    int idx = blockIdx.x * blockDim.x + threadIdx.x;   // L*2*H*H
    if (idx >= L * 2 * H * H) return;
    int k = idx & (H - 1);
    int t1 = idx >> 8;          // /H
    int n = t1 & (H - 1);
    int t2 = t1 >> 8;           // /H
    int t = t2 & 1;
    int l = t2 >> 1;
    int rn = (t == 0) ? 1 : 0;
    int rs = (t == 0) ? 2 : 3;
    size_t in_n = (((size_t)l * 4 + rn) * H + k) * H + n;
    size_t in_s = (((size_t)l * 4 + rs) * H + k) * H + n;
    size_t o    = (((size_t)l * 2 + t) * H + n) * H + k;   // transposed [n][k]
    WlT[o] = f2bf(Wl[in_n]);
    WcT[o] = f2bf(Wr[in_n] + Wl[in_s] + Wr[in_s]);
    if (k == 0)
        biasc[((size_t)l * 2 + t) * H + n] =
            bl[((size_t)l * 4 + rn) * H + n] + bl[((size_t)l * 4 + rs) * H + n];
}

// Transposed bf16 final linear: linT[n*H+k] = lin_W[k*EMB+n], n<EMB
__global__ void prep_lin(const float* __restrict__ lin_W,
                         unsigned short* __restrict__ linT) {
    int idx = blockIdx.x * blockDim.x + threadIdx.x;   // EMB*H
    if (idx >= EMB * H) return;
    int k = idx & (H - 1);
    int n = idx >> 8;
    linT[(size_t)n * H + k] = f2bf(lin_W[(size_t)k * EMB + n]);
}

// Edge scatter-add: 32 lanes per edge, each lane handles 8 contiguous features
// (one 16B bf16 load, 8 f32 atomic adds). Lane 0 bumps the degree counter.
__global__ void scatter_add(const unsigned short* __restrict__ xsrc,
                            const int* __restrict__ src,
                            const int* __restrict__ dst,
                            float* __restrict__ accum,
                            float* __restrict__ cnt) {
    int lane = threadIdx.x & 31;
    int eloc = threadIdx.x >> 5;          // 8 edges per 256-thread block
    int e = blockIdx.x * 8 + eloc;
    if (e >= E) return;
    int s = src[e];
    int d = dst[e];
    const unsigned short* sp = xsrc + (size_t)s * H + lane * 8;
    u16x8 v = *(const u16x8*)sp;
    float* ap = accum + (size_t)d * H + lane * 8;
#pragma unroll
    for (int i = 0; i < 8; ++i) atomicAdd(ap + i, bf2f(v[i]));
    if (lane == 0) atomicAdd(cnt + d, 1.0f);
}

// mean = accum / max(cnt,1), emitted as bf16
__global__ void normalize_bf(const float* __restrict__ accum,
                             const float* __restrict__ cnt,
                             unsigned short* __restrict__ out, int total) {
    int i = blockIdx.x * blockDim.x + threadIdx.x;
    if (i >= total) return;
    float c = fmaxf(cnt[i >> 8], 1.0f);
    out[i] = f2bf(accum[i] / c);
}

// Dual-input WMMA GEMM: C = act( A1@B1t^T + [A2@B2t^T] + bias )
// A: row-major bf16 [M x KK]; B?t: transposed bf16 [N x KK].
// Block = 8 waves owning one 64-column group x 256 rows. The B strips
// (64 x 256 bf16 per matrix, 32KB each) are cooperatively staged into LDS
// once and shared by all 8 waves. Each wave computes a 32x64 strip
// (2 row-tiles x 4 col-tiles = 8 accumulators), so each LDS B fragment
// feeds two back-to-back WMMAs: per k-step 4 A frags + 8 B frags -> 16 WMMAs.
template <bool HAS2, bool RELU, bool OUTBF>
__global__ __launch_bounds__(256) void wmma_gemm(
        const unsigned short* __restrict__ A1,
        const unsigned short* __restrict__ B1t,
        const unsigned short* __restrict__ A2,
        const unsigned short* __restrict__ B2t,
        const float* __restrict__ bias,
        void* __restrict__ Cout,
        int M, int N) {
    constexpr int MATS = HAS2 ? 2 : 1;
    __shared__ unsigned short sB[MATS * 64 * KK];   // [mat][n][k]

    int colGroups = N >> 6;
    int cg = blockIdx.x % colGroups;
    int rowBlock = blockIdx.x / colGroups;
    int col0 = cg << 6;

    // ---- cooperative B staging: MATS * 2048 chunks of 8 ushorts (16B) ----
    {
        const int chunksPerMat = 64 * KK / 8;   // 2048
        for (int i = threadIdx.x; i < MATS * chunksPerMat; i += 256) {
            int mat = HAS2 ? (i >> 11) : 0;     // i / 2048
            int c   = i & (chunksPerMat - 1);
            int n   = c >> 5;                   // KK/8 = 32 chunks per row
            int kc  = (c & 31) * 8;
            const unsigned short* src =
                (mat == 0 ? B1t : B2t) + (size_t)(col0 + n) * KK + kc;
            *(u16x8*)&sB[(mat * 64 + n) * KK + kc] = *(const u16x8*)src;
        }
    }
    __syncthreads();

    int waveId = threadIdx.x >> 5;
    int lane   = threadIdx.x & 31;
    int rowTiles = M >> 4;
    int rt0 = rowBlock * 16 + waveId * 2;       // 2 row-tiles per wave
    if (rt0 >= rowTiles) return;                // wave-uniform; after barrier
    int rt1 = rt0 + 1;
    bool act1 = rt1 < rowTiles;
    int rt1c = act1 ? rt1 : rt0;                // clamp: compute dup, skip store

    int rsel = lane & 15;            // row within tile (A) / col within tile (B)
    int koff = (lane >> 4) << 3;     // K half-split per ISA 16-bit fragment layout

    const unsigned short* a1p0 = A1 + (size_t)((rt0 << 4) + rsel) * KK + koff;
    const unsigned short* a1p1 = A1 + (size_t)((rt1c << 4) + rsel) * KK + koff;
    const unsigned short* a2p0 = HAS2 ? A2 + (size_t)((rt0 << 4) + rsel) * KK + koff : nullptr;
    const unsigned short* a2p1 = HAS2 ? A2 + (size_t)((rt1c << 4) + rsel) * KK + koff : nullptr;
    const unsigned short* b1p[4];
    const unsigned short* b2p[4];
#pragma unroll
    for (int j = 0; j < 4; ++j) {
        b1p[j] = &sB[(j * 16 + rsel) * KK] + koff;
        b2p[j] = HAS2 ? &sB[(64 + j * 16 + rsel) * KK] + koff : nullptr;
    }

    v8f acc[2][4] = {};
#pragma unroll
    for (int k = 0; k < KK; k += 32) {
        v16bf a1_0 = frag16(a1p0 + k);
        v16bf a1_1 = frag16(a1p1 + k);
        v16bf a2_0, a2_1;
        if (HAS2) { a2_0 = frag16(a2p0 + k); a2_1 = frag16(a2p1 + k); }
#pragma unroll
        for (int j = 0; j < 4; ++j) {
            v16bf b1 = frag16(b1p[j] + k);
            acc[0][j] = __builtin_amdgcn_wmma_f32_16x16x32_bf16(
                    false, a1_0, false, b1, (short)0, acc[0][j], false, false);
            acc[1][j] = __builtin_amdgcn_wmma_f32_16x16x32_bf16(
                    false, a1_1, false, b1, (short)0, acc[1][j], false, false);
            if (HAS2) {
                v16bf b2 = frag16(b2p[j] + k);
                acc[0][j] = __builtin_amdgcn_wmma_f32_16x16x32_bf16(
                        false, a2_0, false, b2, (short)0, acc[0][j], false, false);
                acc[1][j] = __builtin_amdgcn_wmma_f32_16x16x32_bf16(
                        false, a2_1, false, b2, (short)0, acc[1][j], false, false);
            }
        }
    }

    // C fragment layout: element v of lane l -> M = v + (l<16?0:8), N = l&15
#pragma unroll
    for (int r = 0; r < 2; ++r) {
        if (r == 1 && !act1) break;             // wave-uniform
        int row0 = ((r == 0 ? rt0 : rt1) << 4);
        int mbase = row0 + ((lane >> 4) << 3);
#pragma unroll
        for (int j = 0; j < 4; ++j) {
            int n = col0 + j * 16 + rsel;
            float bv = bias ? bias[n] : 0.0f;
#pragma unroll
            for (int v = 0; v < 8; ++v) {
                float val = acc[r][j][v] + bv;
                if (RELU) val = fmaxf(val, 0.0f);
                size_t idx = (size_t)(mbase + v) * N + n;
                if (OUTBF) ((unsigned short*)Cout)[idx] = f2bf(val);
                else       ((float*)Cout)[idx] = val;
            }
        }
    }
}

// ---------------- host launch ----------------
extern "C" void kernel_launch(void* const* d_in, const int* in_sizes, int n_in,
                              void* d_out, int out_size, void* d_ws, size_t ws_size,
                              hipStream_t stream) {
    const float* x_a    = (const float*)d_in[0];
    const float* x_b    = (const float*)d_in[1];
    const int*   e_ab   = (const int*)d_in[2];    // [2,E]: row0 src(a), row1 dst(b)
    const int*   e_ba   = (const int*)d_in[3];    // [2,E]: row0 src(b), row1 dst(a)
    const float* Wl     = (const float*)d_in[4];
    const float* bl     = (const float*)d_in[5];
    const float* Wr     = (const float*)d_in[6];
    const float* lin_W  = (const float*)d_in[7];
    const float* lin_b  = (const float*)d_in[8];
    float* out = (float*)d_out;

    char* ws = (char*)d_ws;
    size_t off = 0;
    auto alloc = [&](size_t bytes) -> char* {
        char* p = ws + off;
        off += (bytes + 255) & ~(size_t)255;
        return p;
    };
    const size_t NH = (size_t)NA * H;   // NA==NB

    float* accum_a = (float*)alloc(NH * 4);
    float* accum_b = (float*)alloc(NH * 4);
    float* cnt_a   = (float*)alloc(NA * 4);
    float* cnt_b   = (float*)alloc(NB * 4);
    unsigned short* mean_a = (unsigned short*)alloc(NH * 2);
    unsigned short* mean_b = (unsigned short*)alloc(NH * 2);
    unsigned short* actP0_a = (unsigned short*)alloc(NH * 2);  // ping
    unsigned short* actP0_b = (unsigned short*)alloc(NH * 2);
    unsigned short* actP1_a = (unsigned short*)alloc(NH * 2);  // pong
    unsigned short* actP1_b = (unsigned short*)alloc(NH * 2);
    unsigned short* WlT   = (unsigned short*)alloc((size_t)L * 2 * H * H * 2);
    unsigned short* WcT   = (unsigned short*)alloc((size_t)L * 2 * H * H * 2);
    unsigned short* linT  = (unsigned short*)alloc((size_t)EMB * H * 2);
    float* biasc          = (float*)alloc((size_t)L * 2 * H * 4);

    // weight prep
    prep_weights<<<(L * 2 * H * H) / 256, 256, 0, stream>>>(Wl, bl, Wr, WlT, WcT, biasc);
    prep_lin<<<(EMB * H) / 256, 256, 0, stream>>>(lin_W, linT);

    // activation prep (f32 -> bf16)
    cvt_f32_bf16<<<(int)(NH / 256), 256, 0, stream>>>(x_a, actP0_a, (int)NH);
    cvt_f32_bf16<<<(int)(NH / 256), 256, 0, stream>>>(x_b, actP0_b, (int)NH);

    const int scatterBlocks = (E + 7) / 8;
    const int normBlocks    = (int)(NH / 256);
    const int rowTiles      = NA / 16;                    // 3125
    const int rowBlocksPerCg = (rowTiles + 15) / 16;      // 196 (256 rows/block)
    const int layerBlocks   = rowBlocksPerCg * (H / 64);  // 196*4 = 784
    const int finBlocks     = rowBlocksPerCg * (EMB / 64);// 196*2 = 392

    unsigned short* cur_a = actP0_a; unsigned short* cur_b = actP0_b;
    unsigned short* nxt_a = actP1_a; unsigned short* nxt_b = actP1_b;

    for (int l = 0; l < L; ++l) {
        hipMemsetAsync(accum_a, 0, NH * 4, stream);
        hipMemsetAsync(accum_b, 0, NH * 4, stream);
        hipMemsetAsync(cnt_a,   0, NA * 4, stream);
        hipMemsetAsync(cnt_b,   0, NB * 4, stream);

        // (b,to,a): gather b features into a accumulators
        scatter_add<<<scatterBlocks, 256, 0, stream>>>(cur_b, e_ba, e_ba + E, accum_a, cnt_a);
        // (a,to,b)
        scatter_add<<<scatterBlocks, 256, 0, stream>>>(cur_a, e_ab, e_ab + E, accum_b, cnt_b);

        normalize_bf<<<normBlocks, 256, 0, stream>>>(accum_a, cnt_a, mean_a, (int)NH);
        normalize_bf<<<normBlocks, 256, 0, stream>>>(accum_b, cnt_b, mean_b, (int)NH);

        const unsigned short* wl_a = WlT + ((size_t)l * 2 + 0) * H * H;
        const unsigned short* wc_a = WcT + ((size_t)l * 2 + 0) * H * H;
        const unsigned short* wl_b = WlT + ((size_t)l * 2 + 1) * H * H;
        const unsigned short* wc_b = WcT + ((size_t)l * 2 + 1) * H * H;
        const float* bi_a = biasc + ((size_t)l * 2 + 0) * H;
        const float* bi_b = biasc + ((size_t)l * 2 + 1) * H;

        wmma_gemm<true, true, true><<<layerBlocks, 256, 0, stream>>>(
            mean_a, wl_a, cur_a, wc_a, bi_a, (void*)nxt_a, NA, H);
        wmma_gemm<true, true, true><<<layerBlocks, 256, 0, stream>>>(
            mean_b, wl_b, cur_b, wc_b, bi_b, (void*)nxt_b, NB, H);

        // ping-pong
        unsigned short* t;
        t = cur_a; cur_a = nxt_a; nxt_a = t;
        t = cur_b; cur_b = nxt_b; nxt_b = t;
    }

    // final linear (no relu, f32 out): out_a then out_b concatenated
    wmma_gemm<false, false, false><<<finBlocks, 256, 0, stream>>>(
        cur_a, linT, nullptr, nullptr, lin_b, (void*)out, NA, EMB);
    wmma_gemm<false, false, false><<<finBlocks, 256, 0, stream>>>(
        cur_b, linT, nullptr, nullptr, lin_b, (void*)(out + (size_t)NA * EMB), NB, EMB);
}
